// GlobalAttentionPooling_38173669327083
// MI455X (gfx1250) — compile-verified, matrix operand order checked by reading, not run on hardware
//
#include <hip/hip_runtime.h>
#include <hip/hip_bf16.h>
#include <math.h>

// ---------------------------------------------------------------------------
// GlobalAttentionPooling for MI455X (gfx1250, wave32, WMMA + async-LDS copy)
//   x[N,256] fp32, batch[N] int64 (sorted), Wt[512,256], bt[512], Wg[512], bg
//   out[1024,256] = segment_sum(attn*x) + segment_mean(x)
// ---------------------------------------------------------------------------

typedef __attribute__((ext_vector_type(16))) __bf16 v16bf;
typedef __attribute__((ext_vector_type(8)))  float  v8f;

#define C_DIM  256
#define H_DIM  512
#define B_SEG  1024
#define TILE_M 32

// Monotonic mapping float -> uint32 so unsigned atomicMax == float max.
__device__ __forceinline__ unsigned enc_f32(float f) {
  unsigned u = __float_as_uint(f);
  return (u & 0x80000000u) ? ~u : (u | 0x80000000u);
}
__device__ __forceinline__ float dec_f32(unsigned u) {
  u = (u & 0x80000000u) ? (u & 0x7FFFFFFFu) : ~u;
  return __uint_as_float(u);
}

// ---------------------------------------------------------------------------
// Kernel 0: repack Wt (fp32, [H,C] row-major) into bf16 B-fragments.
// Fragment f = hc*8 + ks covers B-tile (K = ks*32..+32, Nout = hc*16..+16).
// Within a fragment, lane l holds 16 contiguous bf16 (one 32B vector load):
//   element e (j=e/2, p=e&1):  n = hc*16 + (l&15)
//   k = ks*32 + (j<4?0:16) + (l<16?0:8) + (j&3)*2 + p
// (mirrors the documented 16-bit A-matrix 16x32 lane/VGPR striping with the
//  M<->N roles swapped for the B operand).
// ---------------------------------------------------------------------------
__global__ void gap_convert_wt(const float* __restrict__ Wt,
                               __bf16* __restrict__ wtf) {
  int gid  = blockIdx.x * blockDim.x + threadIdx.x;   // 0 .. H*C-1
  int e    = gid & 15;
  int lane = (gid >> 4) & 31;
  int f    = gid >> 9;
  int hc   = f >> 3;
  int ks   = f & 7;
  int j = e >> 1, p = e & 1;
  int n = hc * 16 + (lane & 15);
  int k = ks * 32 + ((j < 4) ? 0 : 16) + ((lane < 16) ? 0 : 8) + ((j & 3) << 1) + p;
  wtf[gid] = (__bf16)Wt[n * C_DIM + k];
}

// ---------------------------------------------------------------------------
// Kernel 1: init logits = bg, zero segment accumulators and d_out.
// ---------------------------------------------------------------------------
__global__ void gap_init(float* __restrict__ logits, const float* __restrict__ bg,
                         unsigned* __restrict__ segmax, float* __restrict__ segsum,
                         float* __restrict__ cnt, float* __restrict__ out,
                         int n, int outsz) {
  int i = blockIdx.x * blockDim.x + threadIdx.x;
  if (i < n)      logits[i] = bg[0];
  if (i < B_SEG)  { segmax[i] = 0u; segsum[i] = 0.0f; cnt[i] = 0.0f; }
  if (i < outsz)  out[i] = 0.0f;
}

// ---------------------------------------------------------------------------
// Kernel 2: fused  logits += sum_h softplus(x @ Wt.T + bt)[.,h] * Wg[h]
// Block = 256 threads = 8 waves, tile = 32 rows.
//   wave -> (msub = wave&1: 16-row subtile, hgrp = wave>>1: H-chunk mod-4)
// x tile is staged into LDS with GLOBAL_LOAD_ASYNC_TO_LDS_B128 (ASYNCcnt),
// bypassing VGPRs entirely; each wave then builds its 8 A-fragments
// (fp32 -> bf16 on gather) and runs 8x v_wmma_f32_16x16x32_bf16 per H-chunk.
// ---------------------------------------------------------------------------
__global__ __launch_bounds__(256, 1)
void gap_gemm_logits(const float* __restrict__ x,
                     const v16bf* __restrict__ wtf,
                     const float* __restrict__ bt,
                     const float* __restrict__ wg,
                     float* __restrict__ logits, int n_rows) {
  __shared__ float sxf[TILE_M * C_DIM];   // 32 KB fp32 x-tile, row-major

  const int tid = threadIdx.x;
  const int m0  = blockIdx.x * TILE_M;

  // Wave-relative LDS byte address of the tile base (addrspace(3) view).
  const unsigned lds_base =
      (unsigned)(size_t)(__attribute__((address_space(3))) float*)sxf;

  // Async global->LDS staging: 16B per issue, no VGPR round-trip.
  for (int ch = tid; ch < TILE_M * C_DIM / 4; ch += 256) {
    int row = ch >> 6;                       // 64 chunks of 16B per row
    int rg  = m0 + row;
    if (rg >= n_rows) rg = n_rows - 1;       // N is a multiple of 32; safety
    const float* gp = &x[(size_t)rg * C_DIM + ((ch & 63) << 2)];
    unsigned lds_off = lds_base + ((unsigned)ch << 4);
    asm volatile("global_load_async_to_lds_b128 %0, %1, off"
                 :: "v"(lds_off), "v"((unsigned long long)(size_t)gp)
                 : "memory");
  }
  asm volatile("s_wait_asynccnt 0x0" ::: "memory");
  __syncthreads();

  const int wave = tid >> 5;
  const int lane = tid & 31;
  const int msub = wave & 1;     // which 16-row subtile
  const int hgrp = wave >> 1;    // H-chunk residue class (mod 4)

  // A fragments: lane l holds row M = msub*16 + (l&15); 16 bf16 per ks.
  v16bf afrag[8];
  const int arow = msub * 16 + (lane & 15);
  #pragma unroll
  for (int ks = 0; ks < 8; ++ks) {
    #pragma unroll
    for (int e = 0; e < 16; ++e) {
      int j = e >> 1, p = e & 1;
      int k = ks * 32 + ((j < 4) ? 0 : 16) + ((lane < 16) ? 0 : 8) + ((j & 3) << 1) + p;
      afrag[ks][e] = (__bf16)sxf[arow * C_DIM + k];
    }
  }

  float partial[8];
  #pragma unroll
  for (int r = 0; r < 8; ++r) partial[r] = 0.0f;

  for (int hc = hgrp; hc < H_DIM / 16; hc += 4) {   // 8 chunks per wave
    v8f acc = {};
    #pragma unroll
    for (int ks = 0; ks < 8; ++ks) {
      v16bf bfrag = wtf[(hc * 8 + ks) * 32 + lane];   // one 32B vector load
      acc = __builtin_amdgcn_wmma_f32_16x16x32_bf16(
          /*neg_a=*/false, afrag[ks], /*neg_b=*/false, bfrag,
          /*c_mod=*/(short)0, acc, /*reuse_a=*/false, /*reuse_b=*/false);
    }
    // Epilogue on the 16x16 tile. C/D layout: VGPR r, lanes 0-15 -> M=r,
    // N=lane; lanes 16-31 -> M=8+r, N=lane-16.
    int   n   = hc * 16 + (lane & 15);
    float btn = bt[n];
    float wgn = wg[n];
    #pragma unroll
    for (int r = 0; r < 8; ++r) {
      float v  = acc[r] + btn;
      float sp = fmaxf(v, 0.0f) + log1pf(expf(-fabsf(v)));  // stable softplus
      partial[r] += sp * wgn;
    }
  }

  // Sum the 16 columns held by each 16-lane half (butterfly within half-wave).
  #pragma unroll
  for (int mask = 1; mask < 16; mask <<= 1) {
    #pragma unroll
    for (int r = 0; r < 8; ++r)
      partial[r] += __shfl_xor(partial[r], mask, 32);
  }

  if ((lane & 15) == 0) {
    // lane 0 -> rows m0+msub*16+0..7, lane 16 -> rows +8..15
    int rbase = m0 + msub * 16 + ((lane >> 4) << 3);
    #pragma unroll
    for (int r = 0; r < 8; ++r) {
      int rg = rbase + r;
      if (rg < n_rows) atomicAdd(&logits[rg], partial[r]);  // 4 H-groups combine
    }
  }
}

// ---------------------------------------------------------------------------
// Kernel 3: segment max of logits (monotonic-uint atomicMax).
// ---------------------------------------------------------------------------
__global__ void gap_segmax(const float* __restrict__ logits,
                           const long long* __restrict__ batch,
                           unsigned* __restrict__ segmax, int n) {
  int i = blockIdx.x * blockDim.x + threadIdx.x;
  if (i >= n) return;
  int b = (int)batch[i];
  atomicMax(&segmax[b], enc_f32(logits[i]));
}

// ---------------------------------------------------------------------------
// Kernel 4: segment sum of exp(logit - max) and segment counts.
// ---------------------------------------------------------------------------
__global__ void gap_segsum(const float* __restrict__ logits,
                           const long long* __restrict__ batch,
                           const unsigned* __restrict__ segmax,
                           float* __restrict__ segsum, float* __restrict__ cnt,
                           int n) {
  int i = blockIdx.x * blockDim.x + threadIdx.x;
  if (i >= n) return;
  int b = (int)batch[i];
  float e = expf(logits[i] - dec_f32(segmax[b]));
  atomicAdd(&segsum[b], e);
  atomicAdd(&cnt[b], 1.0f);
}

// ---------------------------------------------------------------------------
// Kernel 5: per-node coefficient  coef = attn + 1/count  (fuses mean).
// ---------------------------------------------------------------------------
__global__ void gap_coef(const float* __restrict__ logits,
                         const long long* __restrict__ batch,
                         const unsigned* __restrict__ segmax,
                         const float* __restrict__ segsum,
                         const float* __restrict__ cnt,
                         float* __restrict__ coef, int n) {
  int i = blockIdx.x * blockDim.x + threadIdx.x;
  if (i >= n) return;
  int b = (int)batch[i];
  float e = expf(logits[i] - dec_f32(segmax[b]));
  coef[i] = e / segsum[b] + 1.0f / fmaxf(cnt[b], 1.0f);
}

// ---------------------------------------------------------------------------
// Kernel 6: out[b,:] += sum_i coef[i] * x[i,:]   (batch sorted -> run-length
// compression: accumulate in registers while the segment id is unchanged,
// flush with one atomicAdd per (segment, channel) transition).
// ---------------------------------------------------------------------------
__global__ __launch_bounds__(256)
void gap_scatter(const float* __restrict__ x,
                 const long long* __restrict__ batch,
                 const float* __restrict__ coef,
                 float* __restrict__ out, int n_rows) {
  const int c  = threadIdx.x;              // channel (C_DIM == blockDim.x)
  int i0 = blockIdx.x * 256;
  if (i0 >= n_rows) return;
  int i1 = min(i0 + 256, n_rows);

  int   cur_b = (int)batch[i0];
  float acc   = 0.0f;
  for (int i = i0; i < i1; ++i) {
    if (i + 8 < n_rows)
      __builtin_prefetch(&x[(size_t)(i + 8) * C_DIM + c], 0, 1);
    int b = (int)batch[i];
    if (b != cur_b) {
      atomicAdd(&out[(size_t)cur_b * C_DIM + c], acc);
      acc = 0.0f;
      cur_b = b;
    }
    acc += coef[i] * x[(size_t)i * C_DIM + c];
  }
  atomicAdd(&out[(size_t)cur_b * C_DIM + c], acc);
}

// ---------------------------------------------------------------------------
// Host launcher.
// ---------------------------------------------------------------------------
extern "C" void kernel_launch(void* const* d_in, const int* in_sizes, int n_in,
                              void* d_out, int out_size, void* d_ws, size_t ws_size,
                              hipStream_t stream) {
  const float*     x     = (const float*)d_in[0];
  const long long* batch = (const long long*)d_in[1];
  // d_in[2] = batch_size scalar (fixed B_SEG = 1024)
  const float*     Wt    = (const float*)d_in[3];
  const float*     bt    = (const float*)d_in[4];
  const float*     Wg    = (const float*)d_in[5];
  const float*     bg    = (const float*)d_in[6];
  float*           out   = (float*)d_out;

  const int N = in_sizes[1];   // number of nodes

  // Workspace carve-up (all offsets 256B-aligned).
  char*     ws     = (char*)d_ws;
  __bf16*   wtf    = (__bf16*)ws;                           // 256 KB
  size_t    off    = (size_t)H_DIM * C_DIM * sizeof(__bf16);
  float*    logits = (float*)(ws + off); off += (size_t)N * sizeof(float);
  off = (off + 255) & ~(size_t)255;
  unsigned* segmax = (unsigned*)(ws + off); off += B_SEG * sizeof(unsigned);
  float*    segsum = (float*)(ws + off);    off += B_SEG * sizeof(float);
  float*    cnt    = (float*)(ws + off);    off += B_SEG * sizeof(float);
  off = (off + 255) & ~(size_t)255;
  float*    coef   = (float*)(ws + off);    off += (size_t)N * sizeof(float);
  (void)ws_size; (void)n_in;

  const int nb      = (N + 255) / 256;
  const int nb_tile = (N + TILE_M - 1) / TILE_M;

  gap_convert_wt<<<dim3((H_DIM * C_DIM) / 256), dim3(256), 0, stream>>>(Wt, wtf);
  gap_init<<<dim3(nb), dim3(256), 0, stream>>>(logits, bg, segmax, segsum, cnt,
                                               out, N, out_size);
  gap_gemm_logits<<<dim3(nb_tile), dim3(256), 0, stream>>>(
      x, (const v16bf*)wtf, bt, Wg, logits, N);
  gap_segmax<<<dim3(nb), dim3(256), 0, stream>>>(logits, batch, segmax, N);
  gap_segsum<<<dim3(nb), dim3(256), 0, stream>>>(logits, batch, segmax, segsum,
                                                 cnt, N);
  gap_coef<<<dim3(nb), dim3(256), 0, stream>>>(logits, batch, segmax, segsum,
                                               cnt, coef, N);
  gap_scatter<<<dim3(nb), dim3(256), 0, stream>>>(x, batch, coef, out, N);
}
